// WeightedThreeHopGCN_12266426597733
// MI455X (gfx1250) — matrix operand order repaired and checked by reading, not compiled, not run on hardware
//
#include <hip/hip_runtime.h>

#define N_NODES 50000
#define N_EDGES 800000
#define DIM     256
#define F_INS   7
#define KCB     4096

typedef __attribute__((ext_vector_type(16))) __bf16 v16bf;
typedef __attribute__((ext_vector_type(8)))  float  v8f;

union FragB {
    v16bf v;
    uint2 d2[4];
};

// pack two floats into one dword of bf16 pairs (lowers to v_cvt_pk_bf16_f32)
__device__ __forceinline__ unsigned pk_bf16(float a, float b) {
    union { __bf16 h[2]; unsigned u; } t;
    t.h[0] = (__bf16)a;
    t.h[1] = (__bf16)b;
    return t.u;
}

// ---------------- WMMA A fragment (contiguous K order, matches B staging) ----------
// 16x32 bf16 A: lanes 0-15 -> M=lane, K = {k0..k0+15}
//               lanes16-31 -> M=lane-16, K = {k0+16..k0+31}
__device__ __forceinline__ v16bf load_a_frag(const float* __restrict__ row, int k0, bool hi) {
    v16bf a;
    const int kb = k0 + (hi ? 16 : 0);
#pragma unroll
    for (int e = 0; e < 8; ++e) a[e] = (__bf16)row[kb + e];
#pragma unroll
    for (int e = 0; e < 8; ++e) a[8 + e] = (__bf16)row[kb + 8 + e];
    return a;
}

// ---------------- elementwise / reduction helpers ----------------------------

__global__ void zero_f_kernel(float* __restrict__ p, int n) {
    int i = blockIdx.x * blockDim.x + threadIdx.x;
    if (i < n) p[i] = 0.0f;
}

__global__ void maxew_kernel(const float* __restrict__ ew, int e, unsigned* __restrict__ outbits) {
    __shared__ unsigned sm[256];
    unsigned best = 0u;
    for (int i = blockIdx.x * blockDim.x + threadIdx.x; i < e; i += gridDim.x * blockDim.x)
        best = max(best, __float_as_uint(ew[i]));
    sm[threadIdx.x] = best;
    __syncthreads();
    for (int s = 128; s > 0; s >>= 1) {
        if (threadIdx.x < s) sm[threadIdx.x] = max(sm[threadIdx.x], sm[threadIdx.x + s]);
        __syncthreads();
    }
    if (threadIdx.x == 0) atomicMax(outbits, sm[0]);
}

__global__ void degree_kernel(const int* __restrict__ src, const int* __restrict__ dst,
                              int e, float* __restrict__ doutd, float* __restrict__ dind) {
    int i = blockIdx.x * blockDim.x + threadIdx.x;
    if (i < e) {
        atomicAdd(&doutd[src[i]], 1.0f);
        atomicAdd(&dind[dst[i]], 1.0f);
    }
}

__global__ void degfin_kernel(float* __restrict__ p, int n) {
    int i = blockIdx.x * blockDim.x + threadIdx.x;
    if (i < n) p[i] = rsqrtf(fmaxf(p[i], 1.0f));
}

__global__ void lin0_kernel(const float* __restrict__ feat, const float* __restrict__ W0,
                            const float* __restrict__ b0, float* __restrict__ h) {
    int idx = blockIdx.x * blockDim.x + threadIdx.x;
    if (idx >= N_NODES * DIM) return;
    int m = idx >> 8, c = idx & 255;
    float s = b0[c];
#pragma unroll
    for (int j = 0; j < F_INS; ++j) s += feat[m * F_INS + j] * W0[j * DIM + c];
    h[idx] = s;
}

// ---------------- bf16 WMMA GEMM: Out = A[M x 256] @ W[256 x 256] ------------
// Block: 8 waves, tile 128(M) x 64(N). W tile (256K x 64N) staged once into LDS
// in fragment layout [kstep][ntile][epair][lane] (uint2 = 4 bf16 K-values).
// Per k-step: all 4 B fragments are loaded into distinct registers first, then
// 4 WMMAs issue back-to-back (only the first waits on DScnt).
__global__ void gemm_wmma_kernel(const float* __restrict__ A, const float* __restrict__ W,
                                 float* __restrict__ Out, int M) {
    __shared__ uint2 ldsb[8 * 4 * 4 * 32];   // 32 KB: ks*512 + nt*128 + ep*32 + lane
    const int tid = threadIdx.x;
    const int lane = tid & 31;
    const int wid  = tid >> 5;
    const bool hi  = lane >= 16;
    const int ln   = lane & 15;
    const int tileM = blockIdx.x * 128 + wid * 16;
    const int nBase = blockIdx.y * 64;

    // ---- stage W tile (read once per block, coalesced over n) ----
#pragma unroll
    for (int j = 0; j < 16; ++j) {
        int id = tid + 256 * j;              // 4096 uint2 entries
        int l  = id & 31;
        int ep = (id >> 5) & 3;
        int nt = (id >> 7) & 3;
        int ks = id >> 9;
        int n  = nBase + nt * 16 + (l & 15);
        int k  = ks * 32 + ((l >= 16) ? 16 : 0) + ep * 4;
        uint2 w;
        w.x = pk_bf16(W[(k + 0) * DIM + n], W[(k + 1) * DIM + n]);
        w.y = pk_bf16(W[(k + 2) * DIM + n], W[(k + 3) * DIM + n]);
        ldsb[id] = w;
    }
    __syncthreads();

    int rowA = tileM + ln;
    if (rowA >= M) rowA = M - 1;             // clamp; stores guarded
    const float* arow = A + (long)rowA * DIM;

    v8f acc[4] = {};
#pragma unroll
    for (int ks = 0; ks < 8; ++ks) {
        v16bf a = load_a_frag(arow, ks * 32, hi);
        FragB fb[4];                          // distinct regs -> loads overlap WMMAs
#pragma unroll
        for (int nt = 0; nt < 4; ++nt)
#pragma unroll
            for (int ep = 0; ep < 4; ++ep)
                fb[nt].d2[ep] = ldsb[ks * 512 + nt * 128 + ep * 32 + lane];
#pragma unroll
        for (int nt = 0; nt < 4; ++nt)
            acc[nt] = __builtin_amdgcn_wmma_f32_16x16x32_bf16(false, a, false, fb[nt].v,
                                                              (short)0, acc[nt], false, false);
    }

#pragma unroll
    for (int nt = 0; nt < 4; ++nt) {
        const int n = nBase + nt * 16 + ln;
#pragma unroll
        for (int g = 0; g < 8; ++g) {
            int m = tileM + g + (hi ? 8 : 0);
            if (m < M) Out[(long)m * DIM + n] = acc[nt][g];
        }
    }
}

// ---------------- edge scatter: agg[dst] += HW[src] * (dis[src]*ew/ewmax) ----
__global__ void scatter_kernel(const float* __restrict__ HW, const int* __restrict__ src,
                               const int* __restrict__ dst, const float* __restrict__ ew,
                               const float* __restrict__ dis, const unsigned* __restrict__ maxbits,
                               float* __restrict__ agg, int E) {
    const int wid = threadIdx.x >> 5, lane = threadIdx.x & 31;
    const int e = blockIdx.x * 8 + wid;
    if (e >= E) return;
    const float invmax = 1.0f / __uint_as_float(*maxbits);
    const int s = src[e], d = dst[e];
    const float coef = dis[s] * ew[e] * invmax;
    const float* hrow = HW + (long)s * DIM;
    float* arow = agg + (long)d * DIM;
    __builtin_prefetch(hrow, 0, 3);          // global_prefetch_b8
#pragma unroll
    for (int j = 0; j < 8; ++j) {
        int c = j * 32 + lane;
        atomicAdd(&arow[c], hrow[c] * coef);
    }
}

__global__ void finish_kernel(const float* __restrict__ agg, const float* __restrict__ din_is,
                              const float* __restrict__ b, float* __restrict__ h, int relu) {
    int idx = blockIdx.x * blockDim.x + threadIdx.x;
    if (idx >= N_NODES * DIM) return;
    int m = idx >> 8, c = idx & 255;
    float v = agg[idx] * din_is[m] + b[c];
    h[idx] = relu ? fmaxf(v, 0.0f) : v;
}

__global__ void ccnorm_kernel(const float* __restrict__ cb, float* __restrict__ cc) {
    const int wid = threadIdx.x >> 5, lane = threadIdx.x & 31;
    const int code = blockIdx.x * 8 + wid;
    if (code >= KCB) return;
    const float* row = cb + (long)code * DIM;
    float s = 0.0f;
#pragma unroll
    for (int j = 0; j < 8; ++j) { float v = row[j * 32 + lane]; s += v * v; }
#pragma unroll
    for (int off = 16; off; off >>= 1) s += __shfl_xor(s, off, 32);
    if (lane == 0) cc[code] = s;
}

// ---------------- VQ: argmin_n (cc[n] - 2*h.c_n) via bf16 WMMA ---------------
// Block: 8 waves x 16 nodes = 128 rows. A row (8 fragments, 256 K) resident in
// VGPRs. Codebook N-tiles staged into LDS in B-fragment layout, double-buffered
// (2 x 8 KB), shared by all 8 waves; all 8 B fragments of a tile are loaded
// into distinct registers before the 8-WMMA accumulation chain.
__global__ void vq_wmma_kernel(const float* __restrict__ h, const float* __restrict__ cb,
                               const float* __restrict__ cc, float* __restrict__ quant,
                               float* __restrict__ indbuf, float* __restrict__ lossacc, int M) {
    __shared__ uint2 ldsb[2][8 * 4 * 32];    // [buf][ks*128 + ep*32 + lane], 8 KB each
    const int tid = threadIdx.x;
    const int lane = tid & 31;
    const int wid  = tid >> 5;
    const bool hi  = lane >= 16;
    const int ln   = lane & 15;
    const int tileM = blockIdx.x * 128 + wid * 16;

    int rowA = tileM + ln;
    if (rowA >= M) rowA = M - 1;
    const float* arow = h + (long)rowA * DIM;

    v16bf afrag[8];
#pragma unroll
    for (int i = 0; i < 8; ++i) afrag[i] = load_a_frag(arow, i * 32, hi);

    float bestd[8];
    int   bestn[8];
#pragma unroll
    for (int g = 0; g < 8; ++g) { bestd[g] = 3.4e38f; bestn[g] = 0; }

    // stage one 16-code tile: 1024 uint2 entries, 4 per thread, float4 source
    auto stage = [&](int nt, int buf) {
#pragma unroll
        for (int j = 0; j < 4; ++j) {
            int id = tid + 256 * j;
            int l  = id & 31;
            int ep = (id >> 5) & 3;
            int ks = (id >> 7) & 7;
            int n  = nt * 16 + (l & 15);
            int k  = ks * 32 + ((l >= 16) ? 16 : 0) + ep * 4;
            const float4 f = *(const float4*)(cb + (long)n * DIM + k);
            uint2 w;
            w.x = pk_bf16(f.x, f.y);
            w.y = pk_bf16(f.z, f.w);
            ldsb[buf][id] = w;
        }
    };

    stage(0, 0);
    __syncthreads();

    for (int nt = 0; nt < KCB / 16; ++nt) {
        const int buf = nt & 1;
        if (nt + 1 < KCB / 16) stage(nt + 1, 1 - buf);   // overlap with compute

        FragB fb[8];                          // load whole tile's fragments first
#pragma unroll
        for (int ks = 0; ks < 8; ++ks)
#pragma unroll
            for (int ep = 0; ep < 4; ++ep)
                fb[ks].d2[ep] = ldsb[buf][ks * 128 + ep * 32 + lane];

        v8f acc = {};
#pragma unroll
        for (int ks = 0; ks < 8; ++ks)
            acc = __builtin_amdgcn_wmma_f32_16x16x32_bf16(false, afrag[ks], false, fb[ks].v,
                                                          (short)0, acc, false, false);

        const int n = nt * 16 + ln;
        const float ccn = cc[n];
#pragma unroll
        for (int g = 0; g < 8; ++g) {
            float d = ccn - 2.0f * acc[g];               // |h|^2 constant per row -> drop
            if (d < bestd[g]) { bestd[g] = d; bestn[g] = n; }
        }
        __syncthreads();
    }

    // min-reduce across the 16 lanes sharing each M row (tie -> lowest index)
#pragma unroll
    for (int g = 0; g < 8; ++g) {
#pragma unroll
        for (int off = 8; off >= 1; off >>= 1) {
            float od = __shfl_xor(bestd[g], off, 32);
            int   on = __shfl_xor(bestn[g], off, 32);
            if (od < bestd[g] || (od == bestd[g] && on < bestn[g])) { bestd[g] = od; bestn[g] = on; }
        }
    }

    if (ln == 0) {
#pragma unroll
        for (int g = 0; g < 8; ++g) {
            int m = tileM + g + (hi ? 8 : 0);
            if (m < M) indbuf[m] = (float)bestn[g];
        }
    }

    // quantized = codebook[ind] (straight-through == q); commit-loss partials
    float lloss = 0.0f;
#pragma unroll
    for (int r = 0; r < 16; ++r) {
        const int g = r & 7;
        const int srcLane = (r < 8) ? 0 : 16;
        const int ind = __shfl(bestn[g], srcLane, 32);
        const int m = tileM + r;
        if (m < M) {
            const float* crow = cb + (long)ind * DIM;
            const float* hrow = h + (long)m * DIM;
            float* qrow = quant + (long)m * DIM;
#pragma unroll
            for (int j = 0; j < 8; ++j) {
                int c = j * 32 + lane;
                float q = crow[c], hv = hrow[c];
                qrow[c] = q;
                float dd = q - hv;
                lloss += dd * dd;
            }
        }
    }
#pragma unroll
    for (int off = 16; off; off >>= 1) lloss += __shfl_xor(lloss, off, 32);
    if (lane == 0) atomicAdd(lossacc, lloss);
}

__global__ void finloss_kernel(const float* __restrict__ lossacc, float* __restrict__ out) {
    out[0] = lossacc[0] / (float)((long)N_NODES * DIM);
}

// ---------------- launch -----------------------------------------------------

extern "C" void kernel_launch(void* const* d_in, const int* in_sizes, int n_in,
                              void* d_out, int out_size, void* d_ws, size_t ws_size,
                              hipStream_t stream) {
    const float* features = (const float*)d_in[0];
    const int*   src      = (const int*)d_in[1];
    const int*   dst      = (const int*)d_in[2];
    const float* ew       = (const float*)d_in[3];
    const float* W0 = (const float*)d_in[4];
    const float* b0 = (const float*)d_in[5];
    const float* Ws[3] = { (const float*)d_in[6], (const float*)d_in[8], (const float*)d_in[10] };
    const float* bs[3] = { (const float*)d_in[7], (const float*)d_in[9], (const float*)d_in[11] };
    const float* cb = (const float*)d_in[12];

    float* out   = (float*)d_out;
    float* quant = out;                       // [N, D]  (also reused as edge-agg scratch)
    float* hbuf  = out + (long)N_NODES * DIM; // [N, D]
    float* indbf = out + 2L * N_NODES * DIM;  // [N]
    float* lossb = indbf + N_NODES;           // [1]

    float*    ws       = (float*)d_ws;
    unsigned* maxbits  = (unsigned*)ws;       // ws[0]
    float*    lossacc  = ws + 2;
    float*    dout_is  = ws + 16;             // N floats (degree -> rsqrt)
    float*    din_is   = dout_is + N_NODES;   // N floats
    float*    cc       = din_is + N_NODES;    // KCB floats
    float*    tmp      = ws + 104448;         // N*D floats (h @ W scratch)

    const int ND = N_NODES * DIM;
    const int ewBlocks = (N_EDGES + 255) / 256;

    zero_f_kernel<<<(104448 + 255) / 256, 256, 0, stream>>>(ws, 104448);

    maxew_kernel<<<1024, 256, 0, stream>>>(ew, N_EDGES, maxbits);
    degree_kernel<<<ewBlocks, 256, 0, stream>>>(src, dst, N_EDGES, dout_is, din_is);
    degfin_kernel<<<(2 * N_NODES + 255) / 256, 256, 0, stream>>>(dout_is, 2 * N_NODES);

    lin0_kernel<<<(ND + 255) / 256, 256, 0, stream>>>(features, W0, b0, hbuf);

    dim3 ggrid((N_NODES + 127) / 128, DIM / 64, 1);
    for (int l = 0; l < 3; ++l) {
        gemm_wmma_kernel<<<ggrid, 256, 0, stream>>>(hbuf, Ws[l], tmp, N_NODES);
        zero_f_kernel<<<(ND + 255) / 256, 256, 0, stream>>>(quant, ND);  // agg := 0
        scatter_kernel<<<(N_EDGES + 7) / 8, 256, 0, stream>>>(tmp, src, dst, ew, dout_is,
                                                              maxbits, quant, N_EDGES);
        finish_kernel<<<(ND + 255) / 256, 256, 0, stream>>>(quant, din_is, bs[l], hbuf,
                                                            l < 2 ? 1 : 0);
    }

    ccnorm_kernel<<<(KCB + 7) / 8, 256, 0, stream>>>(cb, cc);
    vq_wmma_kernel<<<(N_NODES + 127) / 128, 256, 0, stream>>>(hbuf, cb, cc, quant, indbf,
                                                              lossacc, N_NODES);
    finloss_kernel<<<1, 1, 0, stream>>>(lossacc, lossb);
}
